// MaskedGroupedQuerySelfAttention_34119220199986
// MI455X (gfx1250) — matrix operand order repaired and checked by reading, not compiled, not run on hardware
//
#include <hip/hip_runtime.h>
#include <hip/hip_bf16.h>
#include <math.h>

// ---------------------------------------------------------------------------
// Problem constants (fixed by the reference): B=2, T=2048, D=2048,
// H=16 query heads, KV=4 kv heads, head_dim=128.  M = B*T = 4096.
// All matmul dims divide the tile sizes evenly, so no bounds checks needed.
// ---------------------------------------------------------------------------
#define BB   2
#define TT   2048
#define DD   2048
#define HH   16
#define KVH  4
#define HD   128
#define KVD  (KVH * HD)   // 512
#define MM   (BB * TT)    // 4096

typedef __attribute__((ext_vector_type(16))) _Float16 v16h;
typedef __attribute__((ext_vector_type(8)))  float    v8f;

// ---------------------------------------------------------------------------
// WMMA fragment loaders (wave32); contiguous k order chosen here, same for A and B:
//  A (16x32 f16): lane holds row M = lane&15; K split by lane half:
//    elem e -> K = (lane>>4)*16 + e
//  B (32x16 f16): lane holds col N = lane&15; K rows split by lane half:
//    elem e -> K = (lane>>4)*16 + e
//  C/D (16x16 f32): elem r -> row M = r + 8*(lane>>4), col N = lane&15
// ---------------------------------------------------------------------------
__device__ __forceinline__ v16h load_a_frag(const _Float16* __restrict__ src,
                                            int ld, int lane) {
  const int row = lane & 15;
  const int h16 = (lane >> 4) * 16;
  v16h a;
#pragma unroll
  for (int e = 0; e < 8; ++e) a[e]     = src[row * ld + h16 + e];
#pragma unroll
  for (int e = 0; e < 8; ++e) a[8 + e] = src[row * ld + h16 + 8 + e];
  return a;
}

// B stored row-major as [32 x N], src points at column origin, ld = row stride
__device__ __forceinline__ v16h load_b_frag(const _Float16* __restrict__ src,
                                            int ld, int lane) {
  const int n  = lane & 15;
  const int kb = (lane >> 4) * 16;
  v16h b;
#pragma unroll
  for (int e = 0; e < 16; ++e) b[e] = src[(kb + e) * ld + n];
  return b;
}

// Logical B[k][n] = src[n*ld + k]  (builds K^T fragments from row-major K)
__device__ __forceinline__ v16h load_b_frag_T(const _Float16* __restrict__ src,
                                              int ld, int lane) {
  const int n  = lane & 15;
  const int kb = (lane >> 4) * 16;
  v16h b;
#pragma unroll
  for (int e = 0; e < 16; ++e) b[e] = src[n * ld + kb + e];
  return b;
}

__device__ __forceinline__ v8f wmma_f16(v16h a, v16h b, v8f c) {
  return __builtin_amdgcn_wmma_f32_16x16x32_f16(false, a, false, b,
                                                (short)0, c, false, false);
}

// ---------------------------------------------------------------------------
// f32 -> f16 conversion
// ---------------------------------------------------------------------------
__global__ void cvt_f32_f16(const float* __restrict__ in,
                            _Float16* __restrict__ out, int n) {
  int i = blockIdx.x * blockDim.x + threadIdx.x;
  if (i < n) out[i] = (_Float16)in[i];
}

// ---------------------------------------------------------------------------
// Tiled WMMA GEMM: C[M,N] = A[M,K] * B[K,N], A/B f16 row-major, OutT f16/f32.
// Block: 256 threads (8 waves). Block tile 128x64; wave tile 32x32
// (2x2 of 16x16 WMMA tiles). K-step 32, staged via padded LDS.
// Requires M%128==0, N%64==0, K%32==0.
// ---------------------------------------------------------------------------
template <typename OutT>
__global__ __launch_bounds__(256) void gemm_wmma(
    const _Float16* __restrict__ A, const _Float16* __restrict__ B,
    OutT* __restrict__ C, int M, int N, int K) {
  __shared__ alignas(16) _Float16 As[128 * 40];  // 128x32, pad to 40
  __shared__ alignas(16) _Float16 Bs[32 * 72];   //  32x64, pad to 72

  const int tid  = threadIdx.x;
  const int lane = tid & 31;
  const int wave = tid >> 5;
  const int wm   = (wave & 3) * 32;   // wave M offset in block tile
  const int wn   = (wave >> 2) * 32;  // wave N offset in block tile
  const int mblk = blockIdx.y * 128;
  const int nblk = blockIdx.x * 64;

  v8f acc00 = {}, acc01 = {}, acc10 = {}, acc11 = {};

  for (int k0 = 0; k0 < K; k0 += 32) {
    __syncthreads();
    // A tile 128x32 (4096 halfs): 2 x float4 per thread
#pragma unroll
    for (int j = 0; j < 2; ++j) {
      const int linear = (tid + 256 * j) * 8;
      const int row = linear >> 5, col = linear & 31;
      *(float4*)&As[row * 40 + col] =
          *(const float4*)&A[(size_t)(mblk + row) * K + k0 + col];
    }
    // B tile 32x64 (2048 halfs): 1 x float4 per thread
    {
      const int linear = tid * 8;
      const int row = linear >> 6, col = linear & 63;
      *(float4*)&Bs[row * 72 + col] =
          *(const float4*)&B[(size_t)(k0 + row) * N + nblk + col];
    }
    __syncthreads();

    const v16h a0 = load_a_frag(&As[wm * 40], 40, lane);
    const v16h a1 = load_a_frag(&As[(wm + 16) * 40], 40, lane);
    const v16h b0 = load_b_frag(&Bs[wn], 72, lane);
    const v16h b1 = load_b_frag(&Bs[wn + 16], 72, lane);

    acc00 = wmma_f16(a0, b0, acc00);
    acc01 = wmma_f16(a0, b1, acc01);
    acc10 = wmma_f16(a1, b0, acc10);
    acc11 = wmma_f16(a1, b1, acc11);
  }

  const int ccol = nblk + wn + (lane & 15);
#pragma unroll
  for (int r = 0; r < 8; ++r) {
    const int row0 = mblk + wm + r + 8 * (lane >> 4);
    const int row1 = row0 + 16;
    C[(size_t)row0 * N + ccol]      = (OutT)acc00[r];
    C[(size_t)row0 * N + ccol + 16] = (OutT)acc01[r];
    C[(size_t)row1 * N + ccol]      = (OutT)acc10[r];
    C[(size_t)row1 * N + ccol + 16] = (OutT)acc11[r];
  }
}

// ---------------------------------------------------------------------------
// Flash-attention (causal, GQA). One block = (b, h, 128 query rows).
// 8 waves; wave owns 16 q rows. Keys processed in chunks of 32, K/V chunks
// staged in LDS (shared across waves; same kv head for whole block).
// Q fragments live in registers in A-layout; P restaged via LDS for PV WMMA.
// ---------------------------------------------------------------------------
__global__ __launch_bounds__(256) void attn_kernel(
    const _Float16* __restrict__ Q, const _Float16* __restrict__ Kx,
    const _Float16* __restrict__ Vx, _Float16* __restrict__ Y) {
  __shared__ alignas(16) _Float16 Ks[32 * 128];
  __shared__ alignas(16) _Float16 Vs[32 * 128];
  __shared__ alignas(16) _Float16 Ps[8][16 * 32];

  const int qt   = blockIdx.x;        // query tile: 128 rows
  const int h    = blockIdx.y;        // query head
  const int b    = blockIdx.z;
  const int kvh  = h >> 2;            // repeat_interleave(4): kv head = h/4
  const int tid  = threadIdx.x;
  const int lane = tid & 31;
  const int wave = tid >> 5;
  const int qbase = qt * 128 + wave * 16;
  const int colk  = lane & 15;

  // Preload Q fragments (16 rows x 128 head-dim = 4 A-fragments of 16x32)
  v16h qf[4];
  {
    const int row = qbase + (lane & 15);
    const _Float16* qp = &Q[((size_t)b * TT + row) * DD + h * HD];
    const int h16 = (lane >> 4) * 16;
#pragma unroll
    for (int ks = 0; ks < 4; ++ks) {
#pragma unroll
      for (int e = 0; e < 8; ++e) qf[ks][e]     = qp[ks * 32 + h16 + e];
#pragma unroll
      for (int e = 0; e < 8; ++e) qf[ks][8 + e] = qp[ks * 32 + h16 + 8 + e];
    }
  }

  v8f o[8];
#pragma unroll
  for (int dt = 0; dt < 8; ++dt) o[dt] = (v8f){};
  float m[8], l[8];
#pragma unroll
  for (int r = 0; r < 8; ++r) { m[r] = -INFINITY; l[r] = 0.f; }

  const float scale = 0.08838834764831845f;  // 1/sqrt(128)
  const int nchunks = qt * 4 + 4;            // causal: keys <= qt*128+127

  for (int c = 0; c < nchunks; ++c) {
    const int kc = c * 32;
    __syncthreads();
    // Cooperative load of K/V chunk (32 x 128 halfs each)
#pragma unroll
    for (int j = 0; j < 2; ++j) {
      const int linear = (tid + 256 * j) * 8;
      const int row = linear >> 7, col = linear & 127;
      const size_t g = ((size_t)b * TT + kc + row) * KVD + kvh * HD + col;
      *(float4*)&Ks[row * 128 + col] = *(const float4*)&Kx[g];
      *(float4*)&Vs[row * 128 + col] = *(const float4*)&Vx[g];
    }
    __syncthreads();

    // S = Q * K^T  (16 q x 32 k) via 2 N-subtiles x 4 K-steps
    v8f s0 = {}, s1 = {};
#pragma unroll
    for (int ks = 0; ks < 4; ++ks) {
      const v16h bk0 = load_b_frag_T(&Ks[0 * 128 + ks * 32], 128, lane);
      const v16h bk1 = load_b_frag_T(&Ks[16 * 128 + ks * 32], 128, lane);
      s0 = wmma_f16(qf[ks], bk0, s0);
      s1 = wmma_f16(qf[ks], bk1, s1);
    }

    // scale + causal mask + per-row chunk max
    float mx[8];
#pragma unroll
    for (int r = 0; r < 8; ++r) {
      const int qrow = qbase + r + 8 * (lane >> 4);
      float v0 = s0[r] * scale;
      float v1 = s1[r] * scale;
      if (kc + colk > qrow)      v0 = -INFINITY;
      if (kc + 16 + colk > qrow) v1 = -INFINITY;
      s0[r] = v0; s1[r] = v1;
      mx[r] = fmaxf(v0, v1);
    }
    // row max across 16 lanes of each half-wave (rows live per lane-half)
#pragma unroll
    for (int off = 1; off < 16; off <<= 1)
#pragma unroll
      for (int r = 0; r < 8; ++r)
        mx[r] = fmaxf(mx[r], __shfl_xor(mx[r], off, 32));

    float alpha[8];
#pragma unroll
    for (int r = 0; r < 8; ++r) {
      const float mn = fmaxf(m[r], mx[r]);
      alpha[r] = expf(m[r] - mn);
      m[r] = mn;
    }

    // P = exp(S - m); row sums
    float rs[8];
#pragma unroll
    for (int r = 0; r < 8; ++r) {
      const float p0 = expf(s0[r] - m[r]);
      const float p1 = expf(s1[r] - m[r]);
      s0[r] = p0; s1[r] = p1;
      rs[r] = p0 + p1;
    }
#pragma unroll
    for (int off = 1; off < 16; off <<= 1)
#pragma unroll
      for (int r = 0; r < 8; ++r)
        rs[r] += __shfl_xor(rs[r], off, 32);
#pragma unroll
    for (int r = 0; r < 8; ++r) l[r] = l[r] * alpha[r] + rs[r];
#pragma unroll
    for (int dt = 0; dt < 8; ++dt)
#pragma unroll
      for (int r = 0; r < 8; ++r) o[dt][r] *= alpha[r];

    // Restage P (C/D layout -> LDS row-major -> A-layout fragment)
    _Float16* pw = &Ps[wave][0];
#pragma unroll
    for (int r = 0; r < 8; ++r) {
      const int prow = r + 8 * (lane >> 4);
      pw[prow * 32 + colk]      = (_Float16)s0[r];
      pw[prow * 32 + 16 + colk] = (_Float16)s1[r];
    }
    const v16h pf = load_a_frag(pw, 32, lane);  // LDS ops in-order per wave

    // O += P * V  (16x32 @ 32x128 -> 8 WMMA N-tiles)
#pragma unroll
    for (int dt = 0; dt < 8; ++dt) {
      const v16h vf = load_b_frag(&Vs[dt * 16], 128, lane);
      o[dt] = wmma_f16(pf, vf, o[dt]);
    }
  }

  // Epilogue: Y[b, q, h*128 + d] = O / l
#pragma unroll
  for (int r = 0; r < 8; ++r) {
    const int qrow = qbase + r + 8 * (lane >> 4);
    const float inv = 1.f / l[r];
    _Float16* yp = &Y[((size_t)b * TT + qrow) * DD + h * HD];
#pragma unroll
    for (int dt = 0; dt < 8; ++dt)
      yp[dt * 16 + colk] = (_Float16)(o[dt][r] * inv);
  }
}

// ---------------------------------------------------------------------------
// Launcher. Inputs: x[B,T,D], Wq[D,D], Wk[D,512], Wv[D,512], Wo[D,D] (f32).
// Output: f32 [B,T,D]. Workspace: f16 staging buffers (~80 MB).
// ---------------------------------------------------------------------------
extern "C" void kernel_launch(void* const* d_in, const int* in_sizes, int n_in,
                              void* d_out, int out_size, void* d_ws,
                              size_t ws_size, hipStream_t stream) {
  const float* x  = (const float*)d_in[0];
  const float* Wq = (const float*)d_in[1];
  const float* Wk = (const float*)d_in[2];
  const float* Wv = (const float*)d_in[3];
  const float* Wo = (const float*)d_in[4];
  float* out = (float*)d_out;

  const size_t nX  = (size_t)MM * DD;   // 8388608
  const size_t nWq = (size_t)DD * DD;   // 4194304
  const size_t nWk = (size_t)DD * KVD;  // 1048576
  const size_t nWv = nWk;
  const size_t nWo = nWq;
  const size_t nQ  = nX;
  const size_t nK  = (size_t)MM * KVD;  // 2097152
  const size_t nV  = nK;
  const size_t nA  = nX;

  _Float16* p = (_Float16*)d_ws;
  _Float16* xh  = p;  p += nX;
  _Float16* Wqh = p;  p += nWq;
  _Float16* Wkh = p;  p += nWk;
  _Float16* Wvh = p;  p += nWv;
  _Float16* Woh = p;  p += nWo;
  _Float16* Qh  = p;  p += nQ;
  _Float16* Kh  = p;  p += nK;
  _Float16* Vh  = p;  p += nV;
  _Float16* Ah  = p;  p += nA;
  if (ws_size < (size_t)(p - (_Float16*)d_ws) * sizeof(_Float16)) return;

  const int CT = 256;
  cvt_f32_f16<<<(int)((nX  + CT - 1) / CT), CT, 0, stream>>>(x,  xh,  (int)nX);
  cvt_f32_f16<<<(int)((nWq + CT - 1) / CT), CT, 0, stream>>>(Wq, Wqh, (int)nWq);
  cvt_f32_f16<<<(int)((nWk + CT - 1) / CT), CT, 0, stream>>>(Wk, Wkh, (int)nWk);
  cvt_f32_f16<<<(int)((nWv + CT - 1) / CT), CT, 0, stream>>>(Wv, Wvh, (int)nWv);
  cvt_f32_f16<<<(int)((nWo + CT - 1) / CT), CT, 0, stream>>>(Wo, Woh, (int)nWo);

  // Projections: Q = x@Wq [4096x2048], K/V = x@Wk/Wv [4096x512]
  gemm_wmma<_Float16><<<dim3(DD / 64, MM / 128), 256, 0, stream>>>(
      xh, Wqh, Qh, MM, DD, DD);
  gemm_wmma<_Float16><<<dim3(KVD / 64, MM / 128), 256, 0, stream>>>(
      xh, Wkh, Kh, MM, KVD, DD);
  gemm_wmma<_Float16><<<dim3(KVD / 64, MM / 128), 256, 0, stream>>>(
      xh, Wvh, Vh, MM, KVD, DD);

  // Causal GQA flash attention
  attn_kernel<<<dim3(TT / 128, HH, BB), 256, 0, stream>>>(Qh, Kh, Vh, Ah);

  // Output projection: out = A @ Wo (f32 out)
  gemm_wmma<float><<<dim3(DD / 64, MM / 128), 256, 0, stream>>>(
      Ah, Woh, out, MM, DD, DD);
}
